// Decoder_15324443312437
// MI455X (gfx1250) — matrix operand order repaired and checked by reading, not compiled, not run on hardware
//
#include <hip/hip_runtime.h>
#include <stdint.h>

#define VV 32000
#define HH 1024
#define BB 64
#define TT 64
#define LL 4
#define G4H (4 * HH)

typedef __attribute__((ext_vector_type(16))) __bf16 v16bf;
typedef __attribute__((ext_vector_type(8)))  __bf16 v8bf;
typedef __attribute__((ext_vector_type(8)))  float  v8f;

// ---------- helpers ----------

__device__ __forceinline__ unsigned short f32_to_bf16_bits(float f) {
    unsigned int u = __float_as_uint(f);
    unsigned int r = u + 0x7FFFu + ((u >> 16) & 1u);   // round-to-nearest-even
    return (unsigned short)(r >> 16);
}

__device__ __forceinline__ float sigmoidf_(float x) {
    return 1.0f / (1.0f + __expf(-x));
}

// A fragment: 16x32 bf16, row-major source with leading dim HH.
// lanes 0-15: row m0+lane, K = k0 + {0..15}
// lanes 16-31: row m0+lane-16, K = k0 + {16..31}
__device__ __forceinline__ v16bf load_a_frag(const unsigned short* __restrict__ base,
                                             int m0, int k0, int lane) {
    int row = m0 + (lane & 15);
    int kh  = (lane >> 4) << 4;                 // 0 or 16
    const __bf16* p = reinterpret_cast<const __bf16*>(base) + (size_t)row * HH + k0 + kh;
    v8bf lo = *reinterpret_cast<const v8bf*>(p);
    v8bf hi = *reinterpret_cast<const v8bf*>(p + 8);
    v16bf r;
#pragma unroll
    for (int i = 0; i < 8; ++i) { r[i] = lo[i]; r[i + 8] = hi[i]; }
    return r;
}

// B fragment: 32x16 bf16 of W^T, i.e. B[k][n] = W[n][k], W row-major leading dim HH.
// lane n = colbase + (lane&15); K = k0 + (lane>=16 ? 16 : 0) + {0..15} contiguous in W row.
__device__ __forceinline__ v16bf load_b_frag(const unsigned short* __restrict__ Wcolbase,
                                             int k0, int lane) {
    int col = lane & 15;
    int kb  = k0 + ((lane >> 4) << 4);          // 0 or 16
    const __bf16* p = reinterpret_cast<const __bf16*>(Wcolbase) + (size_t)col * HH + kb;
    return *reinterpret_cast<const v16bf*>(p);
}

// ---------- conversion / gather kernels ----------

__global__ void k_f2bf(const float* __restrict__ src, unsigned short* __restrict__ dst, int n) {
    int i = blockIdx.x * blockDim.x + threadIdx.x;
    if (i < n) dst[i] = f32_to_bf16_bits(src[i]);
}

__global__ void k_gather_x(const int* __restrict__ tgt, const float* __restrict__ emb,
                           unsigned short* __restrict__ xbf) {
    int idx = blockIdx.x * blockDim.x + threadIdx.x;   // [t][b][h]
    if (idx >= TT * BB * HH) return;
    int h = idx & (HH - 1);
    int b = (idx >> 10) & (BB - 1);
    int t = idx >> 16;
    int tok = (t == 0) ? 2 : tgt[b * TT + (t - 1)];
    xbf[idx] = f32_to_bf16_bits(emb[(size_t)tok * HH + h]);
}

// ---------- LSTM step: one layer, one timestep ----------
// grid = 32 blocks x 256 threads = 256 waves; wave -> (m_tile 0..3, j_tile 0..63)
// each wave computes the four 16x16 gate tiles (i,f,g,o) for its (m,j) slice:
// shared A-fragment feeds 4 WMMAs per k-step; cell update done fully in-register.

__global__ void __launch_bounds__(256, 2)
k_lstm_step(const unsigned short* __restrict__ xin,    // [B][H] bf16 (time-t input)
            const unsigned short* __restrict__ hprev,  // [B][H] bf16 (t-1 state)
            const unsigned short* __restrict__ Wi,     // [4H][H] bf16
            const unsigned short* __restrict__ Wh,     // [4H][H] bf16
            const float* __restrict__ bi, const float* __restrict__ bh,  // [4H]
            float* __restrict__ c,                     // [B][H] f32, in-place
            unsigned short* __restrict__ hnext,        // [B][H] bf16
            unsigned short* __restrict__ hfinal_t,     // [B][H] bf16 or nullptr
            float* __restrict__ hout_f32,              // final h slice or nullptr
            float* __restrict__ cout_f32) {            // final c slice or nullptr
    int lane = threadIdx.x & 31;
    int wid  = blockIdx.x * (blockDim.x >> 5) + (threadIdx.x >> 5); // 0..255
    int m0 = (wid >> 6) << 4;     // 0,16,32,48
    int j0 = (wid & 63) << 4;     // 0..1008

    v8f acc[4];
#pragma unroll
    for (int g = 0; g < 4; ++g) acc[g] = (v8f)(0.0f);

    // x @ Wi^T
#pragma unroll 1
    for (int k0 = 0; k0 < HH; k0 += 32) {
        v16bf a = load_a_frag(xin, m0, k0, lane);
#pragma unroll
        for (int g = 0; g < 4; ++g) {
            v16bf b = load_b_frag(Wi + (size_t)(g * HH + j0) * HH, k0, lane);
            acc[g] = __builtin_amdgcn_wmma_f32_16x16x32_bf16(
                false, a, false, b, (short)0, acc[g], false, false);
        }
    }
    // h @ Wh^T
#pragma unroll 1
    for (int k0 = 0; k0 < HH; k0 += 32) {
        v16bf a = load_a_frag(hprev, m0, k0, lane);
#pragma unroll
        for (int g = 0; g < 4; ++g) {
            v16bf b = load_b_frag(Wh + (size_t)(g * HH + j0) * HH, k0, lane);
            acc[g] = __builtin_amdgcn_wmma_f32_16x16x32_bf16(
                false, a, false, b, (short)0, acc[g], false, false);
        }
    }

    int ncol = lane & 15;
    float bias[4];
#pragma unroll
    for (int g = 0; g < 4; ++g)
        bias[g] = bi[g * HH + j0 + ncol] + bh[g * HH + j0 + ncol];

    int rbase = m0 + ((lane >> 4) << 3);   // C/D layout: M = r + 8*(lane>=16)
    int col   = j0 + ncol;
#pragma unroll
    for (int r = 0; r < 8; ++r) {
        int row = rbase + r;
        size_t idx = (size_t)row * HH + col;
        float iv = sigmoidf_(acc[0][r] + bias[0]);
        float fv = sigmoidf_(acc[1][r] + bias[1]);
        float gv = tanhf(acc[2][r] + bias[2]);
        float ov = sigmoidf_(acc[3][r] + bias[3]);
        float cp = c[idx];
        float cn = fv * cp + iv * gv;
        float hn = ov * tanhf(cn);
        c[idx] = cn;
        hnext[idx] = f32_to_bf16_bits(hn);
        if (hfinal_t) hfinal_t[idx] = f32_to_bf16_bits(hn);
        if (hout_f32) { hout_f32[idx] = hn; cout_f32[idx] = cn; }
    }
}

// ---------- Classifier GEMM: [T*B, H] x [V, H]^T + b_cls -> logits ----------
// Register-blocked: each wave computes a 2x4 block of 16x16 tiles (32x64 output).
// Per k-step: 2 A-frags + 4 B-frags feed 8 WMMAs -> ~2.7x less L2 fragment
// traffic than 1 tile/wave. M groups = 128, N groups = 500 -> 64000 waves.

__global__ void __launch_bounds__(256, 2)
k_cls_gemm(const unsigned short* __restrict__ Hf,   // [T*B][H] bf16
           const unsigned short* __restrict__ Wc,   // [V][H] bf16
           const float* __restrict__ bcls,          // [V]
           float* __restrict__ out) {               // [T*B][V]
    const int NGRP = VV / 64;                       // 500
    int lane = threadIdx.x & 31;
    int wid  = blockIdx.x * (blockDim.x >> 5) + (threadIdx.x >> 5);
    int ng = wid % NGRP;
    int mg = wid / NGRP;
    int m0 = mg << 5;        // 32-row block
    int n0 = ng << 6;        // 64-col block

    v8f acc[2][4];
#pragma unroll
    for (int i = 0; i < 2; ++i)
#pragma unroll
        for (int j = 0; j < 4; ++j) acc[i][j] = (v8f)(0.0f);

#pragma unroll 1
    for (int k0 = 0; k0 < HH; k0 += 32) {
        v16bf a0 = load_a_frag(Hf, m0,      k0, lane);
        v16bf a1 = load_a_frag(Hf, m0 + 16, k0, lane);
#pragma unroll
        for (int j = 0; j < 4; ++j) {
            v16bf b = load_b_frag(Wc + (size_t)(n0 + 16 * j) * HH, k0, lane);
            acc[0][j] = __builtin_amdgcn_wmma_f32_16x16x32_bf16(
                false, a0, false, b, (short)0, acc[0][j], false, false);
            acc[1][j] = __builtin_amdgcn_wmma_f32_16x16x32_bf16(
                false, a1, false, b, (short)0, acc[1][j], false, false);
        }
    }

    int ncol = lane & 15;
    int roff = (lane >> 4) << 3;
#pragma unroll
    for (int j = 0; j < 4; ++j) {
        float bias = bcls[n0 + 16 * j + ncol];
#pragma unroll
        for (int i = 0; i < 2; ++i) {
            int rbase = m0 + 16 * i + roff;
            int colj  = n0 + 16 * j + ncol;
#pragma unroll
            for (int r = 0; r < 8; ++r) {
                size_t idx = (size_t)(rbase + r) * VV + colj;
                out[idx] = acc[i][j][r] + bias;
            }
        }
    }
}

// ---------- in-place log-softmax over V per row (online max+sum, 2 passes) ----

__global__ void k_log_softmax(float* __restrict__ out) {
    __shared__ float redm[256];
    __shared__ float reds[256];
    float* row = out + (size_t)blockIdx.x * VV;
    int tid = threadIdx.x;

    // online max + rescaled sum in one sweep
    float m = -3.4e38f, s = 0.0f;
    for (int i = tid; i < VV; i += 256) {
        float x = row[i];
        if (x > m) { s = s * __expf(m - x) + 1.0f; m = x; }
        else       { s += __expf(x - m); }
    }
    redm[tid] = m; reds[tid] = s; __syncthreads();
    for (int st = 128; st > 0; st >>= 1) {
        if (tid < st) {
            float m2 = redm[tid + st], s2 = reds[tid + st];
            float m1 = redm[tid],       s1 = reds[tid];
            float M  = fmaxf(m1, m2);
            redm[tid] = M;
            reds[tid] = s1 * __expf(m1 - M) + s2 * __expf(m2 - M);
        }
        __syncthreads();
    }
    float lse = redm[0] + logf(reds[0]);

    for (int i = tid; i < VV; i += 256) row[i] = row[i] - lse;
}

// ---------- host launcher ----------

extern "C" void kernel_launch(void* const* d_in, const int* in_sizes, int n_in,
                              void* d_out, int out_size, void* d_ws, size_t ws_size,
                              hipStream_t stream) {
    const int*   tgt  = (const int*)  d_in[1];
    const float* h0   = (const float*)d_in[2];
    const float* c0   = (const float*)d_in[3];
    const float* emb  = (const float*)d_in[4];
    const float* W_ih = (const float*)d_in[5];
    const float* W_hh = (const float*)d_in[6];
    const float* b_ih = (const float*)d_in[7];
    const float* b_hh = (const float*)d_in[8];
    const float* Wcls = (const float*)d_in[9];
    const float* bcls = (const float*)d_in[10];

    uint8_t* ws = (uint8_t*)d_ws;
    const size_t nWih  = (size_t)LL * G4H * HH;        // 16.78M elems
    const size_t nWcls = (size_t)VV * HH;              // 32.77M elems
    const size_t nX    = (size_t)TT * BB * HH;         // 4.19M elems
    const size_t nLBH  = (size_t)LL * BB * HH;         // 262K elems

    size_t off = 0;
    unsigned short* Wih_bf  = (unsigned short*)(ws + off); off += nWih  * 2;
    unsigned short* Whh_bf  = (unsigned short*)(ws + off); off += nWih  * 2;
    unsigned short* Wcls_bf = (unsigned short*)(ws + off); off += nWcls * 2;
    unsigned short* x_bf    = (unsigned short*)(ws + off); off += nX    * 2;
    unsigned short* hbuf    = (unsigned short*)(ws + off); off += 2 * nLBH * 2;
    unsigned short* Hfinal  = (unsigned short*)(ws + off); off += nX    * 2;
    float*          c_ws    = (float*)(ws + off);          off += nLBH * 4;

    float* out   = (float*)d_out;
    float* out_h = out + (size_t)TT * BB * VV;
    float* out_c = out_h + nLBH;

    // precision conversion (per call, deterministic)
    k_f2bf<<<(int)((nWih  + 255) / 256), 256, 0, stream>>>(W_ih, Wih_bf,  (int)nWih);
    k_f2bf<<<(int)((nWih  + 255) / 256), 256, 0, stream>>>(W_hh, Whh_bf,  (int)nWih);
    k_f2bf<<<(int)((nWcls + 255) / 256), 256, 0, stream>>>(Wcls, Wcls_bf, (int)nWcls);
    k_f2bf<<<(int)((nLBH  + 255) / 256), 256, 0, stream>>>(h0,   hbuf,    (int)nLBH);
    k_gather_x<<<(int)((nX + 255) / 256), 256, 0, stream>>>(tgt, emb, x_bf);
    hipMemcpyAsync(c_ws, c0, nLBH * sizeof(float), hipMemcpyDeviceToDevice, stream);

    const size_t BH = (size_t)BB * HH;
    for (int t = 0; t < TT; ++t) {
        int p = t & 1;
        const unsigned short* hread  = hbuf + (size_t)p       * nLBH;
        unsigned short*       hwrite = hbuf + (size_t)(1 - p) * nLBH;
        for (int l = 0; l < LL; ++l) {
            const unsigned short* xin =
                (l == 0) ? (x_bf + (size_t)t * BH) : (hwrite + (size_t)(l - 1) * BH);
            k_lstm_step<<<32, 256, 0, stream>>>(
                xin,
                hread + (size_t)l * BH,
                Wih_bf + (size_t)l * G4H * HH,
                Whh_bf + (size_t)l * G4H * HH,
                b_ih + (size_t)l * G4H,
                b_hh + (size_t)l * G4H,
                c_ws + (size_t)l * BH,
                hwrite + (size_t)l * BH,
                (l == LL - 1) ? (Hfinal + (size_t)t * BH) : nullptr,
                (t == TT - 1) ? (out_h + (size_t)l * BH) : nullptr,
                (t == TT - 1) ? (out_c + (size_t)l * BH) : nullptr);
        }
    }

    // classifier: 128 m-groups x 500 n-groups = 64000 waves, 8 waves/block
    k_cls_gemm<<<(128 * (VV / 64)) / 8, 256, 0, stream>>>(Hfinal, Wcls_bf, bcls, out);
    k_log_softmax<<<TT * BB, 256, 0, stream>>>(out);
}
